// Block_28570122453997
// MI455X (gfx1250) — matrix operand order repaired and checked by reading, not compiled, not run on hardware
//
#include <hip/hip_runtime.h>
#include <hip/hip_bf16.h>
#include <math.h>

// ---------------------------------------------------------------------------
// CDNA5 (gfx1250, wave32) implementation of the SE(3) block.
//  - All GEMM-shaped work (convs, MLPs, projections, attention) on
//    v_wmma_f32_16x16x32_f16 with f32 accumulation.
//  - Weights pre-packed once per call into f16 pair-packed (u32) images so
//    the GEMM/conv inner loops are pure uint4 global loads + b128 LDS stores
//    + b32 fragment loads (no per-element cvt/predication).
//  - Probe-pair distance penalty folded into QK^T via a rank-5 augmentation
//    (K = 16 qk dims + 3 probe-dot dims + 2 constant dims = 21 <= 32).
//  - Flash attention with online softmax; S^T layout so softmax reductions
//    are one shfl_xor(16) and P repacks in-register into the B fragment.
// ---------------------------------------------------------------------------

typedef __attribute__((ext_vector_type(16))) _Float16 v16h;
typedef __attribute__((ext_vector_type(8)))  float    v8f;
typedef unsigned int u32;

#define N_SEQ 1024
#define R_ROWS 2048   // B*N

__device__ __forceinline__ float leaky_f(float x){ return x >= 0.f ? x : 0.2f * x; }

__device__ __forceinline__ u32 pack2(float a, float b){
  union { _Float16 h[2]; u32 u; } x;
  x.h[0] = (_Float16)a; x.h[1] = (_Float16)b;
  return x.u;
}

union FragU { v16h v; u32 u[8]; };

__device__ __forceinline__ v8f wmma_f16(v16h a, v16h b, v8f c){
  return __builtin_amdgcn_wmma_f32_16x16x32_f16(false, a, false, b, (short)0, c, false, false);
}

// A-fragment 16x32 from pair-packed LDS (u32 = K pair). Contiguous K order (same as B):
// lane<16 -> K pairs 0..7, lane>=16 -> pairs 8..15
__device__ __forceinline__ v16h load_a_p(const u32* p, int row0, int ldu){
  int lane = threadIdx.x & 31;
  int m  = row0 + (lane & 15);
  int kp = (lane & 16) ? 8 : 0;
  FragU f;
#pragma unroll
  for (int v = 0; v < 8; ++v) f.u[v] = p[m * ldu + kp + v];
  return f.v;
}

// B-fragment 32x16 from pair-packed LDS: lane<16 -> K pairs 0..7, lane>=16 -> 8..15
__device__ __forceinline__ v16h load_b_p(const u32* p, int col0, int ldu){
  int lane = threadIdx.x & 31;
  int c  = col0 + (lane & 15);
  int kp = (lane & 16) ? 8 : 0;
  FragU f;
#pragma unroll
  for (int v = 0; v < 8; ++v) f.u[v] = p[(kp + v) * ldu + c];
  return f.v;
}

// ---------------------------------------------------------------------------
// Weight pre-pack kernels (run once per call; ~3MB total -> negligible at
// 23.3 TB/s, removes all strided f32 gathers + cvt from the WMMA loops).
// ---------------------------------------------------------------------------
// (K,N) f32 row-major -> (K/2, N) u32 with halves (k, k+1)
__global__ void pack_w_kernel(const float* __restrict__ W, u32* __restrict__ WP,
                              int K2, int Nn){
  int i = blockIdx.x * 256 + threadIdx.x;
  if (i < K2 * Nn){
    int p = i / Nn, c = i - p * Nn;
    WP[i] = pack2(W[(size_t)(2*p) * Nn + c], W[(size_t)(2*p+1) * Nn + c]);
  }
}
// conv weight (Cout=256, Cin=256, T=5) -> WP[t][ci/2][co] u32 with halves (ci, ci+1)
__global__ void pack_convw_kernel(const float* __restrict__ W, u32* __restrict__ WP){
  int i = blockIdx.x * 256 + threadIdx.x;   // 5*128*256
  if (i < 163840){
    int t = i / 32768; int rem = i - t * 32768;
    int cp = rem >> 8;  int co = rem & 255;
    WP[i] = pack2(W[(size_t)co * 1280 + (2*cp) * 5 + t],
                  W[(size_t)co * 1280 + (2*cp+1) * 5 + t]);
  }
}

// ---------------------------------------------------------------------------
// Generic WMMA GEMM: C(M,N) = act(A(M,K) @ W(K,N) + bias) [+= C if ACCUM]
// 256 threads = 8 waves; 64x64 tile; K multiple of 32; M multiple of 64.
// W comes pre-packed (K/2, N) u32.
// ---------------------------------------------------------------------------
template<int ACT, int ACCUM>
__global__ void __launch_bounds__(256) gemm_kernel(
    const float* __restrict__ A, const u32* __restrict__ WP,
    const float* __restrict__ bias, float* __restrict__ C,
    int M, int Nn, int K)
{
  __shared__ u32 Asp[64][24];   // 64 rows x 16 K-pairs (+pad)
  __shared__ u32 Wsp[16][68];   // 16 K-pairs x 64 cols (+pad)
  const int m0 = blockIdx.x * 64;
  const int n0 = blockIdx.y * 64;
  const int tid = threadIdx.x;
  const int lane = tid & 31;
  const int wv = tid >> 5;
  const int wr = (wv & 3) * 16;
  const int wc = (wv >> 2) * 32;
  v8f acc0 = {}; v8f acc1 = {};
  for (int k0 = 0; k0 < K; k0 += 32){
    { int r = tid >> 2, cp = (tid & 3) * 4;          // 4 u32 = 8 halfs
      const float4* ap = (const float4*)(A + (size_t)(m0 + r) * K + k0 + cp * 2);
      float4 x = ap[0], y = ap[1];
      uint4 t;
      t.x = pack2(x.x, x.y); t.y = pack2(x.z, x.w);
      t.z = pack2(y.x, y.y); t.w = pack2(y.z, y.w);
      *(uint4*)&Asp[r][cp] = t;
    }
    { int pr = tid >> 4, c4 = (tid & 15) * 4;
      int col = n0 + c4;
      uint4 t;
      if (col + 3 < Nn) t = *(const uint4*)(WP + (size_t)(k0/2 + pr) * Nn + col);
      else { t.x = t.y = t.z = t.w = 0u; }
      *(uint4*)&Wsp[pr][c4] = t;
    }
    __syncthreads();
    v16h af  = load_a_p(&Asp[0][0], wr, 24);
    v16h bf0 = load_b_p(&Wsp[0][0], wc,      68);
    v16h bf1 = load_b_p(&Wsp[0][0], wc + 16, 68);
    acc0 = wmma_f16(af, bf0, acc0);
    acc1 = wmma_f16(af, bf1, acc1);
    __syncthreads();
  }
  const int rb = m0 + wr + ((lane & 16) ? 8 : 0);
#pragma unroll
  for (int half = 0; half < 2; ++half){
    int cc = n0 + wc + half * 16 + (lane & 15);
    if (cc < Nn){
      v8f acc = half ? acc1 : acc0;
#pragma unroll
      for (int r = 0; r < 8; ++r){
        float v = acc[r] + (bias ? bias[cc] : 0.f);
        if (ACT) v = leaky_f(v);
        float* cp = C + (size_t)(rb + r) * Nn + cc;
        if (ACCUM) v += *cp;
        *cp = v;
      }
    }
  }
}

// ---------------------------------------------------------------------------
// Scalar dilated conv (C=256, K=5) as on-the-fly-im2col WMMA GEMM with
// pre-packed weights. Boundary handled at row granularity (no predication
// inside the vector path). Epilogue fuses emb_a + cosine chain-pos embed.
// ---------------------------------------------------------------------------
__global__ void __launch_bounds__(256) conv1d_a_kernel(
    const float* __restrict__ X, const u32* __restrict__ WP,
    const float* __restrict__ bias, const float* __restrict__ extra,
    float* __restrict__ Y, int dil, int addchps)
{
  __shared__ u32 Asp[64][24];
  __shared__ u32 Wsp[16][68];
  const int b  = blockIdx.x >> 4;
  const int n0 = (blockIdx.x & 15) * 64;
  const int c0 = blockIdx.y * 64;
  const int tid = threadIdx.x;
  const int lane = tid & 31;
  const int wv = tid >> 5;
  const int wr = (wv & 3) * 16;
  const int wc = (wv >> 2) * 32;
  v8f acc0 = {}; v8f acc1 = {};
  for (int t = 0; t < 5; ++t){
    const int shift = (t - 2) * dil;
    for (int ci0 = 0; ci0 < 256; ci0 += 32){
      { int r = tid >> 2, cp = (tid & 3) * 4;
        int ns = n0 + r + shift;
        uint4 tv;
        if (ns >= 0 && ns < N_SEQ){
          const float4* xp = (const float4*)(X + ((size_t)(b * N_SEQ + ns)) * 256 + ci0 + cp * 2);
          float4 x = xp[0], y = xp[1];
          tv.x = pack2(x.x, x.y); tv.y = pack2(x.z, x.w);
          tv.z = pack2(y.x, y.y); tv.w = pack2(y.z, y.w);
        } else { tv.x = tv.y = tv.z = tv.w = 0u; }
        *(uint4*)&Asp[r][cp] = tv;
      }
      { int pr = tid >> 4, c4 = (tid & 15) * 4;
        uint4 tv = *(const uint4*)(WP + ((size_t)t * 128 + ci0/2 + pr) * 256 + c0 + c4);
        *(uint4*)&Wsp[pr][c4] = tv;
      }
      __syncthreads();
      v16h af  = load_a_p(&Asp[0][0], wr, 24);
      v16h bf0 = load_b_p(&Wsp[0][0], wc,      68);
      v16h bf1 = load_b_p(&Wsp[0][0], wc + 16, 68);
      acc0 = wmma_f16(af, bf0, acc0);
      acc1 = wmma_f16(af, bf1, acc1);
      __syncthreads();
    }
  }
  const int rn = n0 + wr + ((lane & 16) ? 8 : 0);
#pragma unroll
  for (int half = 0; half < 2; ++half){
    int cc = c0 + wc + half * 16 + (lane & 15);
    v8f acc = half ? acc1 : acc0;
#pragma unroll
    for (int r = 0; r < 8; ++r){
      int nn = rn + r;
      size_t o = ((size_t)(b * N_SEQ + nn)) * 256 + cc;
      float v = acc[r] + bias[cc];
      if (extra)   v += extra[o];
      if (addchps) v += cosf((nn * (1.0f / 1023.0f)) * 3.14159265358979f * (float)(cc + 1));
      Y[o] = v;
    }
  }
}

// ---------------------------------------------------------------------------
// Vector dilated conv (C=64, per-xyz) — VALU with LDS tap staging (<10% FLOPs).
// ---------------------------------------------------------------------------
__global__ void __launch_bounds__(192) conv1d_v_kernel(
    const float* __restrict__ V, const float* __restrict__ Wt,
    const float* __restrict__ extra, float* __restrict__ Y, int dil)
{
  __shared__ float xs[5][192];
  const int row = blockIdx.x;
  const int b = row >> 10, n = row & 1023;
  const int tid = threadIdx.x;
#pragma unroll
  for (int t = 0; t < 5; ++t){
    int ns = n + (t - 2) * dil;
    xs[t][tid] = (ns >= 0 && ns < N_SEQ) ? V[((size_t)(b * N_SEQ + ns)) * 192 + tid] : 0.f;
  }
  __syncthreads();
  int c = tid / 3, x = tid - c * 3;
  float acc = 0.f;
  for (int ci = 0; ci < 64; ++ci){
#pragma unroll
    for (int t = 0; t < 5; ++t) acc += xs[t][ci * 3 + x] * Wt[(size_t)c * 320 + ci * 5 + t];
  }
  if (extra) acc += extra[(size_t)row * 192 + tid];
  Y[(size_t)row * 192 + tid] = acc;
}

// ---------------------------------------------------------------------------
// Edge/node embeddings: first MLP layers (K=4 / K=1) + equivariant embed.
// ---------------------------------------------------------------------------
__global__ void __launch_bounds__(256) embed_kernel(
    const float* __restrict__ pos0, const float* __restrict__ pos1,
    const float* __restrict__ ew0, const float* __restrict__ eb0,
    const float* __restrict__ nw0, const float* __restrict__ nb0,
    const float* __restrict__ eWv, const float* __restrict__ nWv,
    float* __restrict__ hedge, float* __restrict__ hnode, float* __restrict__ embv)
{
  const int row = blockIdx.x;
  const int b = row >> 10, n = row & 1023;
  const int tid = threadIdx.x;
  __shared__ float ev_s[12];   // dn0,dp0,dn1,dp1
  __shared__ float d01_s[3];
  __shared__ float es_s[4];
  __shared__ float ns_s;
  if (tid < 12){
    int e = tid / 3, x = tid - e * 3;
    const float* P = (e < 2) ? pos0 : pos1;
    float v = 0.f;
    if ((e & 1) == 0){ if (n < N_SEQ - 1) v = P[(size_t)(b*N_SEQ + n + 1)*3 + x] - P[(size_t)(b*N_SEQ + n)*3 + x]; }
    else             { if (n > 0)         v = P[(size_t)(b*N_SEQ + n)*3 + x] - P[(size_t)(b*N_SEQ + n - 1)*3 + x]; }
    ev_s[tid] = v;
  }
  if (tid >= 12 && tid < 15){
    int x = tid - 12;
    d01_s[x] = pos1[(size_t)(b*N_SEQ + n)*3 + x] - pos0[(size_t)(b*N_SEQ + n)*3 + x];
  }
  __syncthreads();
  if (tid < 4){
    float s = 1e-8f;
#pragma unroll
    for (int x = 0; x < 3; ++x) s += ev_s[tid*3+x] * ev_s[tid*3+x];
    es_s[tid] = sqrtf(s);
  }
  if (tid == 4){
    float s = 1e-8f;
#pragma unroll
    for (int x = 0; x < 3; ++x) s += d01_s[x] * d01_s[x];
    ns_s = sqrtf(s);
  }
  __syncthreads();
  float he = eb0[tid];
#pragma unroll
  for (int j = 0; j < 4; ++j) he += es_s[j] * ew0[j * 256 + tid];
  hedge[(size_t)row * 256 + tid] = leaky_f(he);
  hnode[(size_t)row * 256 + tid] = leaky_f(nb0[tid] + ns_s * nw0[tid]);
  if (tid < 192){
    int c = tid / 3, x = tid - c * 3;
    float s = nWv[c] * d01_s[x];
#pragma unroll
    for (int j = 0; j < 4; ++j) s += eWv[c * 4 + j] * ev_s[j * 3 + x];
    embv[(size_t)row * 192 + tid] = s;
  }
}

// ---------------------------------------------------------------------------
// Vector side of LocalResidual: lv, v_r1, v_r2, svdot
// ---------------------------------------------------------------------------
__global__ void __launch_bounds__(192) vres1_kernel(
    const float* __restrict__ yv, const float* __restrict__ lrWv,
    const float* __restrict__ Wv1, const float* __restrict__ Wv2,
    float* __restrict__ lvG, float* __restrict__ vr1G, float* __restrict__ svdotG)
{
  __shared__ float yv_s[192];
  __shared__ float v1_s[96];
  __shared__ float v2_s[96];
  const int row = blockIdx.x;
  const int tid = threadIdx.x;
  yv_s[tid] = yv[(size_t)row * 192 + tid];
  __syncthreads();
  { int c = tid / 3, x = tid - c * 3;
    float s = 0.f;
    for (int i = 0; i < 64; ++i) s += lrWv[c * 64 + i] * yv_s[i * 3 + x];
    lvG[(size_t)row * 192 + tid] = s;
  }
  if (tid < 96){
    int r = tid / 3, x = tid - r * 3;
    float a = 0.f, b2 = 0.f;
    for (int i = 0; i < 64; ++i){
      float vv = yv_s[i * 3 + x];
      a  += Wv1[r * 64 + i] * vv;
      b2 += Wv2[r * 64 + i] * vv;
    }
    v1_s[tid] = a; v2_s[tid] = b2;
    vr1G[(size_t)row * 96 + tid] = a;
  }
  __syncthreads();
  if (tid < 32){
    float d = 0.f;
#pragma unroll
    for (int x = 0; x < 3; ++x) d += v1_s[tid * 3 + x] * v2_s[tid * 3 + x];
    svdotG[(size_t)row * 32 + tid] = d;
  }
}

// ---------------------------------------------------------------------------
// z_a = y_a + group_norm(la + p_a); group = 32 channels = exactly one wave32.
// ---------------------------------------------------------------------------
__global__ void __launch_bounds__(256) gnorm_a_kernel(
    const float* __restrict__ la, const float* __restrict__ ya,
    const float* __restrict__ gamma, const float* __restrict__ beta,
    float* __restrict__ za)
{
  const int row = blockIdx.x;
  const int t = threadIdx.x;
  float v = la[(size_t)row * 256 + t];
  float s = v, s2 = v * v;
#pragma unroll
  for (int off = 16; off; off >>= 1){
    s  += __shfl_xor(s,  off, 32);
    s2 += __shfl_xor(s2, off, 32);
  }
  float mu  = s * (1.0f / 32.0f);
  float var = s2 * (1.0f / 32.0f) - mu * mu;
  float nv  = (v - mu) * rsqrtf(var + 1e-5f);
  za[(size_t)row * 256 + t] = ya[(size_t)row * 256 + t] + nv * gamma[t] + beta[t];
}

// ---------------------------------------------------------------------------
// z_v = y_v + vec_group_rms(lv + p_v);  p_v = Wov @ (a_r * v_r1)
// ---------------------------------------------------------------------------
__global__ void __launch_bounds__(192) vres2_kernel(
    const float* __restrict__ lvG, const float* __restrict__ vr1G,
    const float* __restrict__ arG, const float* __restrict__ Wov,
    const float* __restrict__ yv, const float* __restrict__ gnvg,
    float* __restrict__ zv)
{
  __shared__ float ar_s[32];
  __shared__ float v1_s[96];
  __shared__ float g_s[192];
  __shared__ float rms_s[8];
  const int row = blockIdx.x;
  const int tid = threadIdx.x;
  if (tid < 32) ar_s[tid] = arG[(size_t)row * 32 + tid];
  if (tid < 96) v1_s[tid] = vr1G[(size_t)row * 96 + tid];
  __syncthreads();
  int c = tid / 3, x = tid - c * 3;
  float pv = 0.f;
  for (int r = 0; r < 32; ++r) pv += Wov[c * 32 + r] * ar_s[r] * v1_s[r * 3 + x];
  float g = lvG[(size_t)row * 192 + tid] + pv;
  g_s[tid] = g;
  __syncthreads();
  if (tid < 8){
    float s = 0.f;
    for (int k = 0; k < 24; ++k){
      float vv = g_s[tid * 24 + k];
      s += vv * vv;
    }
    rms_s[tid] = sqrtf(s * (1.0f / 8.0f) + 1e-5f);
  }
  __syncthreads();
  zv[(size_t)row * 192 + tid] = yv[(size_t)row * 192 + tid] + g / rms_s[c >> 3] * gnvg[c];
}

// ---------------------------------------------------------------------------
// Probe points + pack augmented QE/KE (f16, K=32, penalty folded) and
// per-head VA/VV in d-major (B,H,16,N) layout for the flash kernel.
// ---------------------------------------------------------------------------
__global__ void __launch_bounds__(256) probe_pack_kernel(
    const float* __restrict__ zv, const float* __restrict__ pos1,
    const float* __restrict__ q, const float* __restrict__ k, const float* __restrict__ va,
    const float* __restrict__ Wvv, const float* __restrict__ pqW, const float* __restrict__ pkW,
    _Float16* __restrict__ QE, _Float16* __restrict__ KE,
    _Float16* __restrict__ VAH, _Float16* __restrict__ VVH)
{
  const int row = blockIdx.x;
  const int b = row >> 10, n = row & 1023;
  const int tid = threadIdx.x;
  __shared__ float zv_s[192];
  __shared__ float vv_s[192];
  __shared__ float pq_s[12], pk_s[12];
  if (tid < 192) zv_s[tid] = zv[(size_t)row * 192 + tid];
  __syncthreads();
  if (tid < 192){
    int c = tid / 3, x = tid - c * 3;
    float s = 0.f;
    for (int i = 0; i < 64; ++i) s += Wvv[c * 64 + i] * zv_s[i * 3 + x];
    vv_s[tid] = s;
  }
  if (tid >= 192 && tid < 216){
    int t = tid - 192;
    int which = t / 12, p = (t % 12) / 3, x = t % 3;
    const float* Wp = which ? pkW : pqW;
    float s = pos1[(size_t)row * 3 + x];
    for (int i = 0; i < 64; ++i) s += Wp[p * 64 + i] * zv_s[i * 3 + x];
    if (which) pk_s[p * 3 + x] = s; else pq_s[p * 3 + x] = s;
  }
  __syncthreads();
  const float r0arr[4] = {0.5f, 1.0f, 2.0f, 4.0f};
  for (int idx = tid; idx < 512; idx += 256){
    int h = idx >> 5, d = idx & 31, p = h & 3;
    float r0 = r0arr[p];
    float inv2 = 1.0f / (r0 * r0);
    float qv = 0.f, kv = 0.f;
    if (d < 16){
      qv = q[(size_t)row * 256 + h * 16 + d] * 0.25f;   // 1/sqrt(DH)
      kv = k[(size_t)row * 256 + h * 16 + d];
    } else if (d < 19){
      int x = d - 16;
      qv = pq_s[p * 3 + x] * inv2;
      kv = pk_s[p * 3 + x];
    } else if (d == 19){
      float s = 0.f;
#pragma unroll
      for (int x = 0; x < 3; ++x) s += pk_s[p*3+x] * pk_s[p*3+x];
      qv = 1.f; kv = -0.5f * s * inv2;
    } else if (d == 20){
      float s = 0.f;
#pragma unroll
      for (int x = 0; x < 3; ++x) s += pq_s[p*3+x] * pq_s[p*3+x];
      qv = -0.5f * s * inv2; kv = 1.f;
    }
    size_t o = ((size_t)(b * 16 + h) * N_SEQ + n) * 32 + d;
    QE[o] = (_Float16)qv; KE[o] = (_Float16)kv;
  }
  {
    int h = tid >> 4, d = tid & 15;   // 256 entries exactly
    size_t o = ((size_t)(b * 16 + h) * 16 + d) * N_SEQ + n;   // d-major
    VAH[o] = (_Float16)va[(size_t)row * 256 + h * 16 + d];
    VVH[o] = (_Float16)((d < 12) ? vv_s[(h * 4 + d / 3) * 3 + (d % 3)] : 0.f);
  }
}

// ---------------------------------------------------------------------------
// Flash attention: S^T = KE @ QE^T (penalty folded in), online softmax,
// O^T = V^T @ P via in-register B-fragment repack of P (one shfl_xor(16)).
// One block = one (b,h), 128 query rows (8 waves x 16). All LDS pair-packed.
// ---------------------------------------------------------------------------
__global__ void __launch_bounds__(256) flash_attn_kernel(
    const _Float16* __restrict__ QE, const _Float16* __restrict__ KE,
    const _Float16* __restrict__ VAH, const _Float16* __restrict__ VVH,
    float* __restrict__ Oa, float* __restrict__ Ov)
{
  const int blk = blockIdx.x;
  const int itile = blk & 7;
  const int bh = blk >> 3;
  const int b = bh >> 4, h = bh & 15;
  const int tid = threadIdx.x;
  const int lane = tid & 31;
  const int wv = tid >> 5;
  const int i_glob = itile * 128 + wv * 16 + (lane & 15);
  const bool hi = (lane & 16) != 0;

  __shared__ u32 kep_s[32][20];    // 32 j-rows x 16 d-pairs (+pad)
  __shared__ u32 vatp_s[16][20];   // 16 d-rows x 16 j-pairs (+pad)
  __shared__ u32 vvtp_s[16][20];

  // QE fragment (B operand), constant across the j loop; pair-packed loads
  v16h qf;
  {
    const u32* qp = (const u32*)(QE + ((size_t)(b * 16 + h) * N_SEQ + i_glob) * 32);
    int kbp = hi ? 8 : 0;
    FragU f;
#pragma unroll
    for (int v = 0; v < 8; ++v) f.u[v] = qp[kbp + v];
    qf = f.v;
  }

  v8f oa = {}; v8f ov = {};
  float m_run = -1e30f, l_run = 0.f;

  const _Float16* keBase = KE  + (size_t)(b * 16 + h) * N_SEQ * 32;
  const _Float16* vaBase = VAH + (size_t)(b * 16 + h) * 16 * N_SEQ;
  const _Float16* vvBase = VVH + (size_t)(b * 16 + h) * 16 * N_SEQ;

  for (int j0 = 0; j0 < N_SEQ; j0 += 32){
    {
      const uint2* keRow = (const uint2*)(keBase + (size_t)j0 * 32);
      uint2 val = keRow[tid];                       // 256 threads x 2 u32 = tile
      int j = (tid * 2) >> 4, dp = (tid * 2) & 15;
      *(uint2*)&kep_s[j][dp] = val;
    }
    {
      int d = tid >> 4, jp = tid & 15;              // 256 u32 per tile
      vatp_s[d][jp] = ((const u32*)(vaBase + (size_t)d * N_SEQ + j0))[jp];
      vvtp_s[d][jp] = ((const u32*)(vvBase + (size_t)d * N_SEQ + j0))[jp];
    }
    if (j0 + 32 < N_SEQ) __builtin_prefetch(keBase + (size_t)(j0 + 32) * 32, 0, 1);
    __syncthreads();

    v16h ka0 = load_a_p(&kep_s[0][0], 0,  20);
    v16h ka1 = load_a_p(&kep_s[0][0], 16, 20);
    v8f z = {};
    v8f st0 = wmma_f16(ka0, qf, z);   // S^T rows j0..15
    v8f st1 = wmma_f16(ka1, qf, z);   // S^T rows j16..31

    float m_t = -1e30f;
#pragma unroll
    for (int r = 0; r < 8; ++r){ m_t = fmaxf(m_t, st0[r]); m_t = fmaxf(m_t, st1[r]); }
    m_t = fmaxf(m_t, __shfl_xor(m_t, 16, 32));
    float m_new = fmaxf(m_run, m_t);
    float scale = __expf(m_run - m_new);
    float p0[8], p1[8];
    float sum = 0.f;
#pragma unroll
    for (int r = 0; r < 8; ++r){
      p0[r] = __expf(st0[r] - m_new);
      p1[r] = __expf(st1[r] - m_new);
      sum += p0[r] + p1[r];
    }
    sum += __shfl_xor(sum, 16, 32);
    l_run = l_run * scale + sum;
    m_run = m_new;
#pragma unroll
    for (int r = 0; r < 8; ++r){ oa[r] *= scale; ov[r] *= scale; }

    // Repack P (f32, S^T register layout) into the B-fragment of P (32j x 16i)
    v16h pf;
#pragma unroll
    for (int r = 0; r < 8; ++r){
      float p0s = __shfl_xor(p0[r], 16, 32);
      float p1s = __shfl_xor(p1[r], 16, 32);
      pf[r]     = (_Float16)(hi ? p1s   : p0[r]);
      pf[8 + r] = (_Float16)(hi ? p1[r] : p0s);
    }

    v16h vaf = load_a_p(&vatp_s[0][0], 0, 20);
    v16h vvf = load_a_p(&vvtp_s[0][0], 0, 20);
    oa = wmma_f16(vaf, pf, oa);
    ov = wmma_f16(vvf, pf, ov);
    __syncthreads();
  }

  float inv_l = 1.0f / l_run;
  const int rbase = hi ? 8 : 0;
#pragma unroll
  for (int r = 0; r < 8; ++r)
    Oa[((size_t)(b * N_SEQ + i_glob)) * 256 + h * 16 + rbase + r] = oa[r] * inv_l;
#pragma unroll
  for (int r = 0; r < 8; ++r){
    int m = rbase + r;
    if (m < 12){
      int dv = m / 3, x = m - dv * 3;
      Ov[(((size_t)(b * N_SEQ + i_glob)) * 64 + h * 4 + dv) * 3 + x] = ov[r] * inv_l;
    }
  }
}

// ---------------------------------------------------------------------------
// Final vector path: z_v_out = z_v + (Wov @ o_v) * sigmoid(|Wov @ o_v|)
// ---------------------------------------------------------------------------
__global__ void __launch_bounds__(192) fin_v_kernel(
    const float* __restrict__ ovG, const float* __restrict__ Wov2,
    const float* __restrict__ zvG, float* __restrict__ outzv)
{
  __shared__ float ov_s[192];
  __shared__ float o2_s[192];
  const int row = blockIdx.x;
  const int tid = threadIdx.x;
  ov_s[tid] = ovG[(size_t)row * 192 + tid];
  __syncthreads();
  int c = tid / 3, x = tid - c * 3;
  float o2 = 0.f;
  for (int i = 0; i < 64; ++i) o2 += Wov2[c * 64 + i] * ov_s[i * 3 + x];
  o2_s[tid] = o2;
  __syncthreads();
  float nrm = sqrtf(o2_s[c*3+0]*o2_s[c*3+0] + o2_s[c*3+1]*o2_s[c*3+1] + o2_s[c*3+2]*o2_s[c*3+2] + 1e-8f);
  float gate = 1.0f / (1.0f + __expf(-nrm));
  outzv[(size_t)row * 192 + tid] = zvG[(size_t)row * 192 + tid] + o2 * gate;
}

__global__ void copy_kernel(float* __restrict__ dst, const float* __restrict__ src, int n){
  int i = blockIdx.x * 256 + threadIdx.x;
  if (i < n) dst[i] = src[i];
}
__global__ void add_kernel(float* __restrict__ dst, const float* __restrict__ a,
                           const float* __restrict__ b, int n){
  int i = blockIdx.x * 256 + threadIdx.x;
  if (i < n) dst[i] = a[i] + b[i];
}

// ---------------------------------------------------------------------------
// Workspace layout (float/u32 slot units)
// ---------------------------------------------------------------------------
#define OFF_HEDGE 0u
#define OFF_HNODE 524288u
#define OFF_EMBA  1048576u
#define OFF_EMBV  1572864u
#define OFF_TA1   1966080u
#define OFF_TA2   2490368u
#define OFF_YA    3014656u
#define OFF_TV1   3538944u
#define OFF_TV2   3932160u
#define OFF_YV    4325376u
#define OFF_LA    4718592u
#define OFF_AR    5242880u
#define OFF_SVDOT 5308416u
#define OFF_LV    5373952u
#define OFF_VR1   5767168u
#define OFF_ZA    5963776u
#define OFF_ZV    6488064u
#define OFF_Q     6881280u
#define OFF_K     7405568u
#define OFF_VAP   7929856u
#define OFF_OA    8454144u
#define OFF_OVV   8978432u
#define OFF_H16   9371648u    // f16 region: QE/KE/VAH/VVH (1572864 float slots)
// pre-packed weight images (u32 slots)
#define PW_EDGE   10944512u
#define PW_NODE   10977280u
#define PW_LR0    11010048u
#define PW_LR1    11042816u
#define PW_LR2    11075584u
#define PW_WQ     11108352u
#define PW_WK     11141120u
#define PW_WVA    11173888u
#define PW_WOA    11206656u
#define PW_SVWA   11239424u
#define PW_SVWOA  11243520u
#define PW_C0     11247616u
#define PW_C1     11411456u
#define PW_C2     11575296u

extern "C" void kernel_launch(void* const* d_in, const int* in_sizes, int n_in,
                              void* d_out, int out_size, void* d_ws, size_t ws_size,
                              hipStream_t stream)
{
  (void)in_sizes; (void)n_in; (void)out_size; (void)ws_size;
  const float* pos0 = (const float*)d_in[0];
  const float* pos1 = (const float*)d_in[1];
  const float* x_a  = (const float*)d_in[2];
  const float* x_v  = (const float*)d_in[3];
  const float* edge_w0 = (const float*)d_in[4];
  const float* edge_b0 = (const float*)d_in[5];
  const float* edge_w1 = (const float*)d_in[6];
  const float* edge_b1 = (const float*)d_in[7];
  const float* edge_Wv = (const float*)d_in[8];
  const float* node_w0 = (const float*)d_in[9];
  const float* node_b0 = (const float*)d_in[10];
  const float* node_w1 = (const float*)d_in[11];
  const float* node_b1 = (const float*)d_in[12];
  const float* node_Wv = (const float*)d_in[13];
  const float* c0aw = (const float*)d_in[14];
  const float* c0ab = (const float*)d_in[15];
  const float* c0vw = (const float*)d_in[16];
  const float* c1aw = (const float*)d_in[17];
  const float* c1ab = (const float*)d_in[18];
  const float* c1vw = (const float*)d_in[19];
  const float* c2aw = (const float*)d_in[20];
  const float* c2ab = (const float*)d_in[21];
  const float* c2vw = (const float*)d_in[22];
  const float* lr_w0 = (const float*)d_in[23];
  const float* lr_b0 = (const float*)d_in[24];
  const float* lr_w1 = (const float*)d_in[25];
  const float* lr_b1 = (const float*)d_in[26];
  const float* lr_w2 = (const float*)d_in[27];
  const float* lr_b2 = (const float*)d_in[28];
  const float* lr_Wv = (const float*)d_in[29];
  const float* sv_Wa = (const float*)d_in[30];
  const float* sv_Wv1 = (const float*)d_in[31];
  const float* sv_Wv2 = (const float*)d_in[32];
  const float* sv_Woa = (const float*)d_in[33];
  const float* sv_Wov = (const float*)d_in[34];
  const float* gna_g = (const float*)d_in[35];
  const float* gna_b = (const float*)d_in[36];
  const float* gnv_g = (const float*)d_in[37];
  const float* pqW = (const float*)d_in[38];
  const float* pkW = (const float*)d_in[39];
  const float* Wq  = (const float*)d_in[40];
  const float* Wk  = (const float*)d_in[41];
  const float* Wva = (const float*)d_in[42];
  const float* Woa = (const float*)d_in[43];
  const float* boa = (const float*)d_in[44];
  const float* Wvv = (const float*)d_in[45];
  const float* Wov = (const float*)d_in[46];

  float* ws = (float*)d_ws;
  float* hedge = ws + OFF_HEDGE;
  float* hnode = ws + OFF_HNODE;
  float* emba  = ws + OFF_EMBA;
  float* embv  = ws + OFF_EMBV;
  float* ta1   = ws + OFF_TA1;
  float* ta2   = ws + OFF_TA2;
  float* ya    = ws + OFF_YA;
  float* tv1   = ws + OFF_TV1;
  float* tv2   = ws + OFF_TV2;
  float* yv    = ws + OFF_YV;
  float* la    = ws + OFF_LA;
  float* ar    = ws + OFF_AR;
  float* svdot = ws + OFF_SVDOT;
  float* lv    = ws + OFF_LV;
  float* vr1   = ws + OFF_VR1;
  float* za    = ws + OFF_ZA;
  float* zv    = ws + OFF_ZV;
  float* qb    = ws + OFF_Q;
  float* kb    = ws + OFF_K;
  float* vab   = ws + OFF_VAP;
  float* oa    = ws + OFF_OA;
  float* ovv   = ws + OFF_OVV;
  _Float16* hbase = (_Float16*)(ws + OFF_H16);
  _Float16* QE  = hbase;
  _Float16* KE  = hbase + 1048576u;
  _Float16* VAH = hbase + 2097152u;
  _Float16* VVH = hbase + 2621440u;
  u32* wsu = (u32*)d_ws;
  u32* pwEdge = wsu + PW_EDGE;   u32* pwNode = wsu + PW_NODE;
  u32* pwLr0  = wsu + PW_LR0;    u32* pwLr1  = wsu + PW_LR1;   u32* pwLr2 = wsu + PW_LR2;
  u32* pwWq   = wsu + PW_WQ;     u32* pwWk   = wsu + PW_WK;    u32* pwWva = wsu + PW_WVA;
  u32* pwWoa  = wsu + PW_WOA;    u32* pwSvWa = wsu + PW_SVWA;  u32* pwSvWoa = wsu + PW_SVWOA;
  u32* pwC0   = wsu + PW_C0;     u32* pwC1   = wsu + PW_C1;    u32* pwC2 = wsu + PW_C2;

  float* out      = (float*)d_out;
  float* out_pos0 = out;
  float* out_pos1 = out + 6144;
  float* out_xaya = out + 12288;
  float* out_xvyv = out + 536576;
  float* out_za   = out + 929792;
  float* out_zv   = out + 1454080;

  const dim3 g64(R_ROWS / 64, 4);     // GEMM N=256
  const dim3 g64n32(R_ROWS / 64, 1);  // GEMM N=32
  const dim3 gconv(32, 4);            // conv_a

  // 0. pre-pack all WMMA-side weights to f16 pair-packed u32 images
  pack_w_kernel<<<128, 256, 0, stream>>>(edge_w1, pwEdge, 128, 256);
  pack_w_kernel<<<128, 256, 0, stream>>>(node_w1, pwNode, 128, 256);
  pack_w_kernel<<<128, 256, 0, stream>>>(lr_w0,  pwLr0, 128, 256);
  pack_w_kernel<<<128, 256, 0, stream>>>(lr_w1,  pwLr1, 128, 256);
  pack_w_kernel<<<128, 256, 0, stream>>>(lr_w2,  pwLr2, 128, 256);
  pack_w_kernel<<<128, 256, 0, stream>>>(Wq,     pwWq,  128, 256);
  pack_w_kernel<<<128, 256, 0, stream>>>(Wk,     pwWk,  128, 256);
  pack_w_kernel<<<128, 256, 0, stream>>>(Wva,    pwWva, 128, 256);
  pack_w_kernel<<<128, 256, 0, stream>>>(Woa,    pwWoa, 128, 256);
  pack_w_kernel<<<16,  256, 0, stream>>>(sv_Wa,  pwSvWa, 128, 32);
  pack_w_kernel<<<16,  256, 0, stream>>>(sv_Woa, pwSvWoa, 16, 256);
  pack_convw_kernel<<<640, 256, 0, stream>>>(c0aw, pwC0);
  pack_convw_kernel<<<640, 256, 0, stream>>>(c1aw, pwC1);
  pack_convw_kernel<<<640, 256, 0, stream>>>(c2aw, pwC2);

  // 1. embeddings (first layers + equivariant embed)
  embed_kernel<<<R_ROWS, 256, 0, stream>>>(pos0, pos1, edge_w0, edge_b0, node_w0, node_b0,
                                           edge_Wv, node_Wv, hedge, hnode, embv);
  // 2. second MLP layers -> emb_a (WMMA)
  gemm_kernel<0,0><<<g64, 256, 0, stream>>>(hedge, pwEdge, edge_b1, emba, R_ROWS, 256, 256);
  gemm_kernel<0,1><<<g64, 256, 0, stream>>>(hnode, pwNode, node_b1, emba, R_ROWS, 256, 256);
  // 3. scalar dilated conv stack (WMMA); last fuses emb_a + cosine pos-embed
  conv1d_a_kernel<<<gconv, 256, 0, stream>>>(x_a, pwC0, c0ab, nullptr, ta1, 4, 0);
  conv1d_a_kernel<<<gconv, 256, 0, stream>>>(ta1, pwC1, c1ab, nullptr, ta2, 2, 0);
  conv1d_a_kernel<<<gconv, 256, 0, stream>>>(ta2, pwC2, c2ab, emba,    ya,  1, 1);
  // 4. vector conv stack; last fuses emb_v
  conv1d_v_kernel<<<R_ROWS, 192, 0, stream>>>(x_v, c0vw, nullptr, tv1, 4);
  conv1d_v_kernel<<<R_ROWS, 192, 0, stream>>>(tv1, c1vw, nullptr, tv2, 2);
  conv1d_v_kernel<<<R_ROWS, 192, 0, stream>>>(tv2, c2vw, embv,    yv,  1);
  // 5. LocalResidual scalar mlp3 (WMMA, conv temps reused)
  gemm_kernel<1,0><<<g64, 256, 0, stream>>>(ya,  pwLr0, lr_b0, ta1, R_ROWS, 256, 256);
  gemm_kernel<1,0><<<g64, 256, 0, stream>>>(ta1, pwLr1, lr_b1, ta2, R_ROWS, 256, 256);
  gemm_kernel<0,0><<<g64, 256, 0, stream>>>(ta2, pwLr2, lr_b2, la,  R_ROWS, 256, 256);
  // 6. vector residual pieces; p_a folded into la via accumulate-GEMM
  vres1_kernel<<<R_ROWS, 192, 0, stream>>>(yv, lr_Wv, sv_Wv1, sv_Wv2, lv, vr1, svdot);
  gemm_kernel<0,0><<<g64n32, 256, 0, stream>>>(ya, pwSvWa, nullptr, ar, R_ROWS, 32, 256);
  gemm_kernel<0,1><<<g64, 256, 0, stream>>>(svdot, pwSvWoa, nullptr, la, R_ROWS, 256, 32);
  // 7. norms -> z_a, z_v
  gnorm_a_kernel<<<R_ROWS, 256, 0, stream>>>(la, ya, gna_g, gna_b, za);
  vres2_kernel<<<R_ROWS, 192, 0, stream>>>(lv, vr1, ar, sv_Wov, yv, gnv_g, zv);
  // 8. attention projections (WMMA)
  gemm_kernel<0,0><<<g64, 256, 0, stream>>>(za, pwWq,  nullptr, qb,  R_ROWS, 256, 256);
  gemm_kernel<0,0><<<g64, 256, 0, stream>>>(za, pwWk,  nullptr, kb,  R_ROWS, 256, 256);
  gemm_kernel<0,0><<<g64, 256, 0, stream>>>(za, pwWva, nullptr, vab, R_ROWS, 256, 256);
  // 9. probes + augmented f16 packing (penalty folded into K=32 contraction)
  probe_pack_kernel<<<R_ROWS, 256, 0, stream>>>(zv, pos1, qb, kb, vab, Wvv, pqW, pkW,
                                                QE, KE, VAH, VVH);
  // 10. flash attention (WMMA, online softmax)
  flash_attn_kernel<<<256, 256, 0, stream>>>(QE, KE, VAH, VVH, oa, ovv);
  // 11. outputs
  copy_kernel<<<(6144 + 255) / 256, 256, 0, stream>>>(out_pos0, pos0, 6144);
  copy_kernel<<<(6144 + 255) / 256, 256, 0, stream>>>(out_pos1, pos1, 6144);
  add_kernel<<<(524288 + 255) / 256, 256, 0, stream>>>(out_xaya, x_a, ya, 524288);
  add_kernel<<<(393216 + 255) / 256, 256, 0, stream>>>(out_xvyv, x_v, yv, 393216);
  copy_kernel<<<(524288 + 255) / 256, 256, 0, stream>>>(out_za, za, 524288);
  gemm_kernel<0,1><<<g64, 256, 0, stream>>>(oa, pwWoa, boa, out_za, R_ROWS, 256, 256);
  fin_v_kernel<<<R_ROWS, 192, 0, stream>>>(ovv, Wov, zv, out_zv);
}
